// Codebook_54752243089494
// MI455X (gfx1250) — matrix operand order repaired and checked by reading, not compiled, not run on hardware
//
#include <hip/hip_runtime.h>
#include <hip/hip_bf16.h>

// ---------------------------------------------------------------------------
// Problem constants (from reference): z [B=32, C=256, H=64, W=64], K=1024.
// N = B*H*W = 131072 rows of dim D = 256.
// ---------------------------------------------------------------------------
#define BATCH   32
#define DMODEL  256
#define HW      4096           // 64*64
#define NROWS   131072         // BATCH*HW
#define KCODES  1024
#define DECAY_F 0.99f
#define OMD_F   0.01f          // 1 - DECAY
#define EPS_F   1e-7f

// WMMA tiling
#define KCHUNK    256          // codes staged in LDS per pass (4 passes)
#define LDS_PITCH 264          // bf16 elems per LDS row: 528B, 16B aligned, bank-conflict-free
#define LDS_BYTES (KCHUNK * LDS_PITCH * 2)   // 135168 B (dynamic LDS, WGP has 320KB)

typedef __attribute__((ext_vector_type(16))) __bf16 v16bf;
typedef __attribute__((ext_vector_type(8)))  float  v8f;

__device__ __forceinline__ unsigned short f2bf_bits(float f) {
    __bf16 h = (__bf16)f;
    unsigned short u;
    __builtin_memcpy(&u, &h, 2);
    return u;
}

// ---------------------------------------------------------------------------
// K0: zero a float region of scratch
// ---------------------------------------------------------------------------
__global__ void zero_kernel(float* __restrict__ p, int n) {
    int i = blockIdx.x * blockDim.x + threadIdx.x;
    if (i < n) p[i] = 0.0f;
}

// ---------------------------------------------------------------------------
// K1/K5: generic batched 32x32 LDS-tiled transpose.
// For each z-slice (blockIdx.z): out[c*R + r] = in[r*C + c],  in is [R][C].
// block = (32, 8)
// ---------------------------------------------------------------------------
__global__ void transpose2d(const float* __restrict__ in, float* __restrict__ out,
                            int R, int C) {
    __shared__ float tile[32][33];
    size_t base = (size_t)blockIdx.z * (size_t)R * (size_t)C;
    int r0 = blockIdx.y * 32, c0 = blockIdx.x * 32;
    int tx = threadIdx.x, ty = threadIdx.y;
#pragma unroll
    for (int i = 0; i < 32; i += 8) {
        tile[ty + i][tx] = in[base + (size_t)(r0 + ty + i) * C + (c0 + tx)];
    }
    __syncthreads();
#pragma unroll
    for (int i = 0; i < 32; i += 8) {
        out[base + (size_t)(c0 + ty + i) * R + (r0 + tx)] = tile[tx][ty + i];
    }
}

// ---------------------------------------------------------------------------
// K2: e_sq[k] = sum_c embedding[k][c]^2.  One block (256 thr) per code.
// ---------------------------------------------------------------------------
__global__ void esq_kernel(const float* __restrict__ emb, float* __restrict__ esq) {
    __shared__ float red[8];
    int k = blockIdx.x;
    int t = threadIdx.x;
    float v = emb[(size_t)k * DMODEL + t];
    v *= v;
#pragma unroll
    for (int off = 16; off > 0; off >>= 1) v += __shfl_xor(v, off, 32);
    if ((t & 31) == 0) red[t >> 5] = v;
    __syncthreads();
    if (t == 0) {
        float s = 0.0f;
#pragma unroll
        for (int i = 0; i < 8; ++i) s += red[i];
        esq[k] = s;
    }
}

// ---------------------------------------------------------------------------
// K3: WMMA distance + argmin kernel.
//   block = 256 threads (8 waves); each wave owns 16 rows -> 128 rows/block.
//   grid.x = NROWS/128 = 1024.
//   A (16x32 bf16) fragments for the full d=256 kept in registers (a[8]).
//   embedding staged into LDS as bf16 in KCHUNK-code passes; each wave walks
//   16 column tiles per pass. The 8-step k-accumulation is split into TWO
//   independent WMMA chains so the scheduler can overlap ds_load_b128 of one
//   chain with v_wmma of the other (halves serial dependency depth).
//   dist = e_sq[col] - 2*dot  (row norm constant -> irrelevant for argmin);
//   e_sq values for the pass are hoisted into registers.
// ---------------------------------------------------------------------------
__global__ __launch_bounds__(256)
void vq_argmin_wmma(const float* __restrict__ enc,     // [NROWS][256]
                    const float* __restrict__ emb,     // [1024][256]
                    const float* __restrict__ esq,     // [1024]
                    int*   __restrict__ out_idx,       // [NROWS] (scratch, int)
                    float* __restrict__ out_idx_f) {   // [NROWS] (d_out #3)
    extern __shared__ unsigned short smem[];           // [KCHUNK][LDS_PITCH] bf16

    const int tid   = threadIdx.x;
    const int lane  = tid & 31;
    const int wave  = tid >> 5;                        // 0..7
    const int m     = lane & 15;                       // row-in-tile / col-in-tile
    const int khalf = lane >> 4;                       // 0 or 1
    const int rowBase = blockIdx.x * 128 + wave * 16;

    // ---- Load A fragments for this wave's 16 rows, full d=256 (8 k-chunks).
    // A operand in contiguous k order (B uses the same): lane m (half h): VGPR v -> k = 16*h + 2*v + {0,1}
    v16bf a[8];
    const float2* arow = (const float2*)(enc + (size_t)(rowBase + m) * DMODEL);
#pragma unroll
    for (int kc = 0; kc < 8; ++kc) {
#pragma unroll
        for (int v = 0; v < 8; ++v) {
            int k0 = kc * 32 + (khalf << 4) + (v << 1);
            float2 f = arow[k0 >> 1];
            a[kc][2 * v]     = (__bf16)f.x;
            a[kc][2 * v + 1] = (__bf16)f.y;
        }
    }

    float best[8];
    int   bidx[8];
#pragma unroll
    for (int r = 0; r < 8; ++r) { best[r] = 3.4e38f; bidx[r] = 0; }

    for (int kb = 0; kb < KCODES; kb += KCHUNK) {
        __syncthreads();   // protect LDS reuse across passes
        // ---- Stage KCHUNK codes of embedding into LDS as bf16 (coalesced fp32 reads).
        for (int idx = tid; idx < (KCHUNK * DMODEL) / 2; idx += 256) {
            int lin  = idx << 1;
            int code = lin >> 8;          // / DMODEL
            int d0   = lin & (DMODEL - 1);
            float2 f = *(const float2*)(emb + (size_t)(kb + code) * DMODEL + d0);
            unsigned int pk = (unsigned int)f2bf_bits(f.x) |
                              ((unsigned int)f2bf_bits(f.y) << 16);
            *(unsigned int*)(smem + code * LDS_PITCH + d0) = pk;
        }

        // ---- Hoist this lane's e_sq values for the whole pass (16 tiles).
        float es[KCHUNK / 16];
#pragma unroll
        for (int ct = 0; ct < KCHUNK / 16; ++ct) {
            es[ct] = esq[kb + ct * 16 + m];
        }
        __syncthreads();

#pragma unroll 1
        for (int ct = 0; ct < KCHUNK / 16; ++ct) {
            const int cbase = ct * 16;
            // B operand in the same contiguous k order as A: lane n=m, half h: 16 contiguous k = kc*32 + 16h .. +15
            const unsigned short* brow = smem + (cbase + m) * LDS_PITCH + (khalf << 4);
            // Two independent accumulator chains (kc even vs kc odd halves)
            // -> two live B buffers, overlapped ds_load / v_wmma issue.
            v8f acc0 = {};
            v8f acc1 = {};
#pragma unroll
            for (int kc = 0; kc < 4; ++kc) {
                v16bf b0 = *(const v16bf*)(brow + (2 * kc) * 32);
                v16bf b1 = *(const v16bf*)(brow + (2 * kc + 1) * 32);
                acc0 = __builtin_amdgcn_wmma_f32_16x16x32_bf16(
                        false, a[2 * kc], false, b0, (short)0, acc0, false, false);
                acc1 = __builtin_amdgcn_wmma_f32_16x16x32_bf16(
                        false, a[2 * kc + 1], false, b1, (short)0, acc1, false, false);
            }
            const int col = kb + cbase + m;                 // this lane's column
#pragma unroll
            for (int r = 0; r < 8; ++r) {
                float dot = acc0[r] + acc1[r];
                float dv  = __builtin_fmaf(-2.0f, dot, es[ct]);
                if (dv < best[r]) { best[r] = dv; bidx[r] = col; }
            }
        }
    }

    // ---- Argmin reduce across the 16 lanes that share each row.
    // (xor masks 1..8 stay within lane halves; C layout: VGPR r -> row r + 8*khalf)
#pragma unroll
    for (int off = 8; off > 0; off >>= 1) {
#pragma unroll
        for (int r = 0; r < 8; ++r) {
            float ov = __shfl_xor(best[r], off, 32);
            int   oi = __shfl_xor(bidx[r], off, 32);
            if (ov < best[r] || (ov == best[r] && oi < bidx[r])) {
                best[r] = ov; bidx[r] = oi;
            }
        }
    }
    if ((lane & 15) == 0) {
        int rowOff = rowBase + (khalf << 3);
#pragma unroll
        for (int r = 0; r < 8; ++r) {
            out_idx[rowOff + r]   = bidx[r];
            out_idx_f[rowOff + r] = (float)bidx[r];
        }
    }
}

// ---------------------------------------------------------------------------
// K4: gather codebook rows into quantized_flat + segment sums via L2 atomics.
//     One block (256 thr) per row.
// ---------------------------------------------------------------------------
__global__ void gather_stats(const float* __restrict__ enc,
                             const float* __restrict__ emb,
                             const int*   __restrict__ idx,
                             float* __restrict__ qflat,     // [NROWS][256]
                             float* __restrict__ segsum,    // [K][256]
                             float* __restrict__ counts) {  // [K]
    int n = blockIdx.x;
    int c = threadIdx.x;
    int k = idx[n];
    float e = emb[(size_t)k * DMODEL + c];
    qflat[(size_t)n * DMODEL + c] = e;
    atomicAdd(&segsum[(size_t)k * DMODEL + c], enc[(size_t)n * DMODEL + c]);
    if (c == 0) atomicAdd(&counts[k], 1.0f);
}

// ---------------------------------------------------------------------------
// K6: EMA usage + global sum n (for Laplace normalization).
// ---------------------------------------------------------------------------
__global__ void usage_kernel(const float* __restrict__ usage,
                             const float* __restrict__ counts,
                             float* __restrict__ raw,
                             float* __restrict__ nsum) {
    __shared__ float red[8];
    int k = blockIdx.x * 256 + threadIdx.x;
    float v = usage[k] * DECAY_F + counts[k] * OMD_F;
    raw[k] = v;
#pragma unroll
    for (int off = 16; off > 0; off >>= 1) v += __shfl_xor(v, off, 32);
    if ((threadIdx.x & 31) == 0) red[threadIdx.x >> 5] = v;
    __syncthreads();
    if (threadIdx.x == 0) {
        float s = 0.0f;
#pragma unroll
        for (int i = 0; i < 8; ++i) s += red[i];
        atomicAdd(nsum, s);
    }
}

// ---------------------------------------------------------------------------
// K7: final codebook EMA outputs. One block (256 thr) per code.
// ---------------------------------------------------------------------------
__global__ void finalize_kernel(const float* __restrict__ code_avg,
                                const float* __restrict__ segsum,
                                const float* __restrict__ raw,
                                const float* __restrict__ nsum,
                                float* __restrict__ new_emb,
                                float* __restrict__ new_usage,
                                float* __restrict__ new_avg) {
    int k = blockIdx.x;
    int c = threadIdx.x;
    float n  = *nsum;
    float nu = (raw[k] + EPS_F) / (n + (float)KCODES * EPS_F) * n;
    if (c == 0) new_usage[k] = nu;
    size_t off = (size_t)k * DMODEL + c;
    float na = code_avg[off] * DECAY_F + OMD_F * segsum[off];
    new_avg[off] = na;
    new_emb[off] = na / nu;
}

// ---------------------------------------------------------------------------
// Launcher
// ---------------------------------------------------------------------------
extern "C" void kernel_launch(void* const* d_in, const int* in_sizes, int n_in,
                              void* d_out, int out_size, void* d_ws, size_t ws_size,
                              hipStream_t stream) {
    const float* z        = (const float*)d_in[0];   // [32,256,64,64]
    const float* emb      = (const float*)d_in[1];   // [1024,256]
    const float* usage    = (const float*)d_in[2];   // [1024]
    const float* code_avg = (const float*)d_in[3];   // [1024,256]

    float* out = (float*)d_out;
    // Output regions (concatenated flat in reference return order, fp32):
    float* o_enc  = out;                       // encoded_flat   [N,256]
    float* o_qf   = out + 33554432;            // quantized_flat [N,256]
    float* o_idxf = out + 67108864;            // codebook_indices [N] (as f32)
    float* o_q    = out + 67239936;            // quantized [B,256,H,W]
    float* o_nemb = out + 100794368;           // new_embedding [K,256]
    float* o_nuse = out + 101056512;           // new_code_usage [K]
    float* o_navg = out + 101057536;           // new_code_avg [K,256]

    // Workspace layout (float slots):
    float* wsf       = (float*)d_ws;
    int*   ws_idx    = (int*)d_ws;             // [N] int           (131072 slots)
    float* ws_esq    = wsf + 131072;           // [K]
    float* ws_counts = wsf + 132096;           // [K]
    float* ws_segsum = wsf + 133120;           // [K*256]
    float* ws_raw    = wsf + 395264;           // [K]
    float* ws_nsum   = wsf + 396288;           // [1]

    // K0: zero counts + segsum + raw + nsum (contiguous): 264193 floats
    {
        int nz = 264193;
        zero_kernel<<<(nz + 255) / 256, 256, 0, stream>>>(ws_counts, nz);
    }

    // K1: z [B][256][4096] -> encoded_flat [B][4096][256]
    {
        dim3 grid(HW / 32, DMODEL / 32, BATCH);   // (128, 8, 32)
        dim3 block(32, 8);
        transpose2d<<<grid, block, 0, stream>>>(z, o_enc, DMODEL, HW);
    }

    // K2: e_sq
    esq_kernel<<<KCODES, 256, 0, stream>>>(emb, ws_esq);

    // K3: WMMA distances + argmin
    {
        dim3 grid(NROWS / 128);                   // 1024 blocks
        vq_argmin_wmma<<<grid, 256, LDS_BYTES, stream>>>(
            o_enc, emb, ws_esq, ws_idx, o_idxf);
    }

    // K4: gather + segment sums
    gather_stats<<<NROWS, 256, 0, stream>>>(o_enc, emb, ws_idx,
                                            o_qf, ws_segsum, ws_counts);

    // K5: quantized_flat [B][4096][256] -> quantized [B][256][4096]
    {
        dim3 grid(DMODEL / 32, HW / 32, BATCH);   // (8, 128, 32)
        dim3 block(32, 8);
        transpose2d<<<grid, block, 0, stream>>>(o_qf, o_q, HW, DMODEL);
    }

    // K6: EMA usage + total n
    usage_kernel<<<KCODES / 256, 256, 0, stream>>>(usage, ws_counts, ws_raw, ws_nsum);

    // K7: finalize codebook outputs
    finalize_kernel<<<KCODES, 256, 0, stream>>>(code_avg, ws_segsum, ws_raw,
                                                ws_nsum, o_nemb, o_nuse, o_navg);
}
